// DistributedGridSelfAttentionCpp_62182536511492
// MI455X (gfx1250) — matrix operand order repaired and checked by reading, not compiled, not run on hardware
//
#include <hip/hip_runtime.h>

typedef __attribute__((ext_vector_type(16))) _Float16 v16h;
typedef __attribute__((ext_vector_type(8)))  _Float16 v8h;
typedef __attribute__((ext_vector_type(4)))  _Float16 v4h;
typedef __attribute__((ext_vector_type(8)))  float    v8f;
typedef __attribute__((ext_vector_type(4)))  float    v4f;

#define NDIM 384
#define CDIM 128
#define HDIM 4
#define DDIM 32
#define NPIX (NDIM * NDIM)  // 147456

union V16 { v16h v; v8h h[2]; };

__device__ __forceinline__ v8f wmma_f16(v16h a, v16h b, v8f c) {
  // D = A(16x32 f16) * B(32x16 f16) + C(16x16 f32)
  return __builtin_amdgcn_wmma_f32_16x16x32_f16(false, a, false, b, (short)0, c,
                                                false, false);
}

// Low 32 bits of a generic pointer to __shared__ == LDS byte offset
// (flat aperture).
__device__ __forceinline__ unsigned lds_off(const void* p) {
  return (unsigned)(unsigned long long)p;
}

// CDNA5 async DMA: 16 bytes per lane, global -> LDS, tracked by ASYNCcnt.
__device__ __forceinline__ void async_copy_b128(unsigned ldsoff, const void* gsrc) {
  asm volatile("global_load_async_to_lds_b128 %0, %1, off"
               :: "v"(ldsoff), "v"(gsrc)
               : "memory");
}
__device__ __forceinline__ void wait_async0() {
  asm volatile("s_wait_asynccnt 0x0" ::: "memory");
}

// ---------------------------------------------------------------------------
// Kernel 1: convert weights to f16. Layout in ws (f16 elements):
//   wq@0, wk@16384, wv@32768, wg@49152, wo@65536, wbias(16x128, zero pad)@81920
// ---------------------------------------------------------------------------
__global__ void k_prep(const float* __restrict__ wq, const float* __restrict__ wk,
                       const float* __restrict__ wv, const float* __restrict__ wg,
                       const float* __restrict__ wo, const float* __restrict__ bw,
                       _Float16* __restrict__ o) {
  int id = blockIdx.x * 256 + threadIdx.x;
  if (id < 5 * 16384) {
    int w = id >> 14, off = id & 16383;
    const float* src = (w == 0) ? wq : (w == 1) ? wk : (w == 2) ? wv
                        : (w == 3) ? wg : wo;
    o[id] = (_Float16)src[off];
  } else {
    int bid = id - 5 * 16384;  // 0..2047
    int r = bid >> 7, c = bid & 127;
    o[5 * 16384 + bid] = (_Float16)((r < HDIM) ? bw[r * CDIM + c] : 0.0f);
  }
}

// ---------------------------------------------------------------------------
// Kernel 2: LayerNorm over C=128 per pixel. One wave (32 lanes x 4 ch) / pixel.
// ---------------------------------------------------------------------------
__global__ void k_ln(const float* __restrict__ pair, const float* __restrict__ lnw,
                     const float* __restrict__ lnb, _Float16* __restrict__ xo) {
  int wave = threadIdx.x >> 5, lane = threadIdx.x & 31;
  size_t pix = (size_t)blockIdx.x * 8 + wave;
  v4f v = *(const v4f*)(pair + pix * CDIM + lane * 4);
  float s = v[0] + v[1] + v[2] + v[3];
  float q = v[0] * v[0] + v[1] * v[1] + v[2] * v[2] + v[3] * v[3];
#pragma unroll
  for (int off = 16; off > 0; off >>= 1) {
    s += __shfl_xor(s, off);
    q += __shfl_xor(q, off);
  }
  float mean = s * (1.0f / CDIM);
  float var = q * (1.0f / CDIM) - mean * mean;
  float rs = rsqrtf(var + 1e-5f);
  v4h out;
#pragma unroll
  for (int e = 0; e < 4; e++) {
    float w = lnw[lane * 4 + e], b = lnb[lane * 4 + e];
    out[e] = (_Float16)((v[e] - mean) * rs * w + b);
  }
  *(v4h*)(xo + pix * CDIM + lane * 4) = out;
}

// ---------------------------------------------------------------------------
// Kernel 3: fused projections. Block = 256 thr (8 waves) handles 16 pixels.
// Wave w owns output cols [16w,16w+16) of Q,K,V,Gate; wave 0 also pair bias.
// Y = X @ W^T : A = X tile (async-staged LDS), B[k][n] = W[n][k].
// ---------------------------------------------------------------------------
__global__ void k_proj(const _Float16* __restrict__ x,
                       const _Float16* __restrict__ wqm, const _Float16* __restrict__ wkm,
                       const _Float16* __restrict__ wvm, const _Float16* __restrict__ wgm,
                       const _Float16* __restrict__ wbm,
                       _Float16* __restrict__ qb, _Float16* __restrict__ kb,
                       _Float16* __restrict__ vb, _Float16* __restrict__ gb,
                       float* __restrict__ biasb) {
  __shared__ alignas(64) _Float16 lx[16 * CDIM];
  int t = threadIdx.x;
  size_t m0 = (size_t)blockIdx.x * 16;
  // async stage 16x128 f16 tile: 16 bytes per thread
  async_copy_b128(lds_off(&lx[(t >> 4) * CDIM + (t & 15) * 8]),
                  x + (m0 + (t >> 4)) * CDIM + (t & 15) * 8);
  wait_async0();
  __syncthreads();

  int lane = t & 31, w = t >> 5;
  int ln = lane & 15, half = lane >> 4;
  int n0 = w * 16;
  __builtin_prefetch(wqm + (size_t)n0 * CDIM, 0, 1);
  __builtin_prefetch(wkm + (size_t)n0 * CDIM, 0, 1);
  __builtin_prefetch(wvm + (size_t)n0 * CDIM, 0, 1);
  __builtin_prefetch(wgm + (size_t)n0 * CDIM, 0, 1);

  v8f aq = {}, ak = {}, av = {}, ag = {}, ab = {};
#pragma unroll
  for (int k0 = 0; k0 < CDIM; k0 += 32) {
    V16 a;
    a.h[0] = *(const v8h*)&lx[ln * CDIM + k0 + half * 16];
    a.h[1] = *(const v8h*)&lx[ln * CDIM + k0 + half * 16 + 8];
    int wrow = (n0 + ln) * CDIM + k0 + half * 16;
    v16h Bq = *(const v16h*)(wqm + wrow);
    v16h Bk = *(const v16h*)(wkm + wrow);
    v16h Bv = *(const v16h*)(wvm + wrow);
    v16h Bg = *(const v16h*)(wgm + wrow);
    aq = wmma_f16(a.v, Bq, aq);
    ak = wmma_f16(a.v, Bk, ak);
    av = wmma_f16(a.v, Bv, av);
    ag = wmma_f16(a.v, Bg, ag);
    if (w == 0) {  // wave-uniform branch: EXEC stays all-ones
      v16h Bb = *(const v16h*)(wbm + ln * CDIM + k0 + half * 16);
      ab = wmma_f16(a.v, Bb, ab);
    }
  }
#pragma unroll
  for (int r = 0; r < 8; r++) {
    size_t p = m0 + r + 8 * half;
    int c = n0 + ln;
    qb[p * CDIM + c] = (_Float16)aq[r];
    kb[p * CDIM + c] = (_Float16)ak[r];
    vb[p * CDIM + c] = (_Float16)av[r];
    gb[p * CDIM + c] = (_Float16)(1.0f / (1.0f + __expf(-ag[r])));
  }
  if (w == 0 && ln < HDIM) {
#pragma unroll
    for (int r = 0; r < 8; r++)
      biasb[(size_t)ln * NPIX + (m0 + r + 8 * half)] = ab[r];
  }
}

// ---------------------------------------------------------------------------
// Kernel 4: flash attention. Block = 256 thr (8 waves), one pair-row b and a
// 32-query i-tile; wave w handles head (w&3), query subtile (w>>2).
// Loops 12 j-tiles of 32 keys; Q/K tiles async-DMA'd to LDS.
// ---------------------------------------------------------------------------
__global__ void k_attn(const _Float16* __restrict__ qbuf, const _Float16* __restrict__ kbuf,
                       const _Float16* __restrict__ vbuf, const _Float16* __restrict__ gbuf,
                       const float* __restrict__ biasb, const float* __restrict__ mask,
                       _Float16* __restrict__ gobuf) {
  __shared__ alignas(64) _Float16 lds_q[32 * CDIM];    // 8 KB
  __shared__ alignas(64) _Float16 lds_k[32 * CDIM];    // 8 KB
  __shared__ alignas(64) _Float16 lds_vt[CDIM * 32];   // 8 KB (transposed)
  __shared__ alignas(64) _Float16 lds_p[8][16 * 32];   // 8 KB

  const int b = blockIdx.x;
  const int i0 = blockIdx.y * 32;
  const int t = threadIdx.x;
  const int lane = t & 31, w = t >> 5;
  const int hh = w & 3, is = w >> 2;
  const int ln = lane & 15, half = lane >> 4;
  const float scale = 0.17677669529663687f;  // 1/sqrt(32)

  // async stage Q tile (32x128): 16 bytes per thread
  async_copy_b128(lds_off(&lds_q[(t >> 3) * CDIM + (t & 7) * 16]),
                  qbuf + ((size_t)(b * NDIM + i0 + (t >> 3))) * CDIM + (t & 7) * 16);
  wait_async0();
  __syncthreads();

  // A operand for Q (built once): row m=ln of this wave's query subtile
  V16 aqv;
  aqv.h[0] = *(const v8h*)&lds_q[(is * 16 + ln) * CDIM + hh * DDIM + half * 16];
  aqv.h[1] = *(const v8h*)&lds_q[(is * 16 + ln) * CDIM + hh * DDIM + half * 16 + 8];

  float mrun[8], lrun[8];
  v8f acco[2] = {{}, {}};
#pragma unroll
  for (int r = 0; r < 8; r++) { mrun[r] = -3e38f; lrun[r] = 0.0f; }

  for (int jt = 0; jt < 12; jt++) {
    const int j0 = jt * 32;
    __syncthreads();  // previous-iteration LDS reads done
    // async stage K tile (32x128): 16 bytes per thread
    async_copy_b128(lds_off(&lds_k[(t >> 3) * CDIM + (t & 7) * 16]),
                    kbuf + ((size_t)(b * NDIM + j0 + (t >> 3))) * CDIM + (t & 7) * 16);
    // stage V transposed: thread <-> (channel c = t&127, j-half = t>>7)
    {
      int c = t & 127, jh = t >> 7;
      v16h col;
#pragma unroll
      for (int jj = 0; jj < 16; jj++)
        col[jj] = vbuf[((size_t)(b * NDIM + j0 + jh * 16 + jj)) * CDIM + c];
      *(v16h*)&lds_vt[c * 32 + jh * 16] = col;
    }
    wait_async0();
    __syncthreads();

    // S = Q K^T for two 16-col subtiles
    float lg[2][8];
#pragma unroll
    for (int sub = 0; sub < 2; sub++) {
      v16h bk = *(const v16h*)&lds_k[(sub * 16 + ln) * CDIM + hh * DDIM + half * 16];
      v8f z = {};
      v8f s = wmma_f16(aqv.v, bk, z);
      float mb = 1e9f * (mask[b * NDIM + j0 + sub * 16 + ln] - 1.0f);
#pragma unroll
      for (int r = 0; r < 8; r++) {
        float bia = biasb[(size_t)hh * NPIX +
                          (size_t)(i0 + is * 16 + r + 8 * half) * NDIM +
                          j0 + sub * 16 + ln];
        lg[sub][r] = s[r] * scale + bia + mb;
      }
    }
    // online softmax (rows live in 16-lane groups)
    float alpha[8], p0[8], p1[8];
#pragma unroll
    for (int r = 0; r < 8; r++) {
      float tm = fmaxf(lg[0][r], lg[1][r]);
#pragma unroll
      for (int off = 8; off > 0; off >>= 1) tm = fmaxf(tm, __shfl_xor(tm, off));
      float mnew = fmaxf(mrun[r], tm);
      alpha[r] = __expf(mrun[r] - mnew);
      mrun[r] = mnew;
      p0[r] = __expf(lg[0][r] - mnew);
      p1[r] = __expf(lg[1][r] - mnew);
      float ts = p0[r] + p1[r];
#pragma unroll
      for (int off = 8; off > 0; off >>= 1) ts += __shfl_xor(ts, off);
      lrun[r] = lrun[r] * alpha[r] + ts;
    }
    // rescale accumulators, stash P to LDS in row-major
#pragma unroll
    for (int r = 0; r < 8; r++) {
      acco[0][r] *= alpha[r];
      acco[1][r] *= alpha[r];
      int m = r + 8 * half;
      lds_p[w][m * 32 + ln] = (_Float16)p0[r];
      lds_p[w][m * 32 + 16 + ln] = (_Float16)p1[r];
    }
    // O += P @ V  (K-dim = 32 keys, two 16-wide d tiles)
    V16 ap;
    ap.h[0] = *(const v8h*)&lds_p[w][ln * 32 + half * 16];
    ap.h[1] = *(const v8h*)&lds_p[w][ln * 32 + half * 16 + 8];
#pragma unroll
    for (int dt = 0; dt < 2; dt++) {
      v16h bv = *(const v16h*)&lds_vt[(hh * DDIM + dt * 16 + ln) * 32 + half * 16];
      acco[dt] = wmma_f16(ap.v, bv, acco[dt]);
    }
  }

  // normalize, gate, store go (f16)
#pragma unroll
  for (int r = 0; r < 8; r++) {
    float inv = 1.0f / lrun[r];
    size_t pix = (size_t)(b * NDIM + i0 + is * 16 + r + 8 * half);
#pragma unroll
    for (int dt = 0; dt < 2; dt++) {
      int c = hh * DDIM + dt * 16 + ln;
      float g = (float)gbuf[pix * CDIM + c];
      gobuf[pix * CDIM + c] = (_Float16)(acco[dt][r] * inv * g);
    }
  }
}

// ---------------------------------------------------------------------------
// Kernel 5: out = go @ o_w^T, f32 result
// ---------------------------------------------------------------------------
__global__ void k_out(const _Float16* __restrict__ go, const _Float16* __restrict__ wom,
                      float* __restrict__ out) {
  __shared__ alignas(64) _Float16 lx[16 * CDIM];
  int t = threadIdx.x;
  size_t m0 = (size_t)blockIdx.x * 16;
  async_copy_b128(lds_off(&lx[(t >> 4) * CDIM + (t & 15) * 8]),
                  go + (m0 + (t >> 4)) * CDIM + (t & 15) * 8);
  wait_async0();
  __syncthreads();

  int lane = t & 31, w = t >> 5;
  int ln = lane & 15, half = lane >> 4;
  int n0 = w * 16;
  __builtin_prefetch(wom + (size_t)n0 * CDIM, 0, 1);
  v8f acc = {};
#pragma unroll
  for (int k0 = 0; k0 < CDIM; k0 += 32) {
    V16 a;
    a.h[0] = *(const v8h*)&lx[ln * CDIM + k0 + half * 16];
    a.h[1] = *(const v8h*)&lx[ln * CDIM + k0 + half * 16 + 8];
    v16h B = *(const v16h*)(wom + (n0 + ln) * CDIM + k0 + half * 16);
    acc = wmma_f16(a.v, B, acc);
  }
#pragma unroll
  for (int r = 0; r < 8; r++)
    out[(m0 + r + 8 * half) * CDIM + n0 + ln] = acc[r];
}

// ---------------------------------------------------------------------------
extern "C" void kernel_launch(void* const* d_in, const int* in_sizes, int n_in,
                              void* d_out, int out_size, void* d_ws, size_t ws_size,
                              hipStream_t stream) {
  const float* pair = (const float*)d_in[0];
  const float* mask = (const float*)d_in[1];
  const float* ln_w = (const float*)d_in[2];
  const float* ln_b = (const float*)d_in[3];
  const float* bias_w = (const float*)d_in[4];
  const float* q_w = (const float*)d_in[5];
  const float* k_w = (const float*)d_in[6];
  const float* v_w = (const float*)d_in[7];
  const float* g_w = (const float*)d_in[8];
  const float* o_w = (const float*)d_in[9];

  char* ws = (char*)d_ws;
  // f16 weights block (k_prep fills element ids 0..83967 from ws base)
  _Float16* wall = (_Float16*)ws;
  _Float16* wq = wall + 0;
  _Float16* wk = wall + 16384;
  _Float16* wv = wall + 32768;
  _Float16* wg = wall + 49152;
  _Float16* wo = wall + 65536;
  _Float16* wb = wall + 81920;  // 16x128 zero-padded bias_w

  const size_t SZX = (size_t)NPIX * CDIM * sizeof(_Float16);  // 37748736 B
  size_t off = 262144;  // 256 KB reserved for weights
  _Float16* xh = (_Float16*)(ws + off); off += SZX;
  _Float16* qh = (_Float16*)(ws + off); off += SZX;
  _Float16* kh = (_Float16*)(ws + off); off += SZX;
  _Float16* vh = (_Float16*)(ws + off); off += SZX;
  _Float16* gh = (_Float16*)(ws + off); off += SZX;
  float* biasb = (float*)(ws + off);    off += (size_t)HDIM * NPIX * sizeof(float);
  _Float16* goh = (_Float16*)(ws + off);

  k_prep<<<328, 256, 0, stream>>>(q_w, k_w, v_w, g_w, o_w, bias_w, wall);
  k_ln<<<NPIX / 8, 256, 0, stream>>>(pair, ln_w, ln_b, xh);
  k_proj<<<NPIX / 16, 256, 0, stream>>>(xh, wq, wk, wv, wg, wb, qh, kh, vh, gh, biasb);
  dim3 ga(NDIM, NDIM / 32);
  k_attn<<<ga, 256, 0, stream>>>(qh, kh, vh, gh, biasb, mask, goh);
  k_out<<<NPIX / 16, 256, 0, stream>>>(goh, wo, (float*)d_out);
}
